// NewModel_42356967473589
// MI455X (gfx1250) — matrix operand order repaired and checked by reading, not compiled, not run on hardware
//
#include <hip/hip_runtime.h>

// ---------------------------------------------------------------------------
// Shapes (fixed by the reference)
// ---------------------------------------------------------------------------
#define BB 4
#define CC 2048
#define QQ 64
#define DD 1024
#define NEGV 1e30f

typedef float v8f  __attribute__((ext_vector_type(8)));
typedef __bf16 v16bf __attribute__((ext_vector_type(16)));
typedef unsigned short us8 __attribute__((ext_vector_type(8)));
typedef int v4i __attribute__((ext_vector_type(4)));

union Frag16 { v16bf v; us8 h[2]; };

__device__ __forceinline__ unsigned short f2bf(float f) {
  union { float f; unsigned int u; } c; c.f = f;
  unsigned int r = c.u + 0x7FFFu + ((c.u >> 16) & 1u);   // round-to-nearest-even
  return (unsigned short)(r >> 16);
}

// --------------------------- async copy support ----------------------------
#if defined(__has_builtin)
#if __has_builtin(__builtin_amdgcn_global_load_async_to_lds_b128)
#define HAVE_ASYNC_LDS 1
#endif
#endif

#if defined(__has_builtin)
#if __has_builtin(__builtin_amdgcn_s_wait_asynccnt)
#define WAITASYNC(n) __builtin_amdgcn_s_wait_asynccnt(n)
#else
#define WAITASYNC(n) asm volatile("s_wait_asynccnt %0" ::"i"(n))
#endif
#else
#define WAITASYNC(n) asm volatile("s_wait_asynccnt %0" ::"i"(n))
#endif

typedef __attribute__((address_space(1))) v4i gas_v4i;   // global int4
typedef __attribute__((address_space(3))) v4i las_v4i;   // LDS int4

// ---------------------------------------------------------------------------
// Generic wave-tile bf16 WMMA GEMM (for the small GEMMs):
//   C[m,n] = sum_k A[m,k] * Bt[n,k]  (+epilogue)
// mode 0: plain   mode 2: +rowb[m]+colb[n]-NEG*(1-mask[n])
// ---------------------------------------------------------------------------
__global__ void gemm_bf16_kernel(const unsigned short* __restrict__ A,
                                 const unsigned short* __restrict__ Bt,
                                 float* __restrict__ Co,
                                 int M, int N, int K,
                                 long long sA, long long sB, long long sC,
                                 int batches, int mode,
                                 const float* __restrict__ rowb,
                                 const float* __restrict__ colb,
                                 const int*   __restrict__ maskp) {
  int wid  = (int)(((size_t)blockIdx.x * blockDim.x + threadIdx.x) >> 5);
  int lane = threadIdx.x & 31;
  int mtiles = M >> 4, ngroups = N >> 6;
  int per_batch = mtiles * ngroups;
  if (wid >= per_batch * batches) return;
  int b  = wid / per_batch;
  int t  = wid - b * per_batch;
  int m0 = (t / ngroups) << 4;
  int n0 = (t % ngroups) << 6;
  A  += (long long)b * sA;
  Bt += (long long)b * sB;
  Co += (long long)b * sC;
  if (mode == 2) { rowb += (long long)b * M; colb += (long long)b * N; maskp += (long long)b * N; }

  int low = lane & 15, hi = lane >> 4;
  v8f acc[4] = {};

  const unsigned short* arow = A + (size_t)(m0 + low) * K + 8 * hi;
  for (int k0 = 0; k0 < K; k0 += 32) {
    Frag16 af;
    af.h[0] = *(const us8*)(arow + k0);
    af.h[1] = *(const us8*)(arow + k0 + 16);
#pragma unroll
    for (int j = 0; j < 4; ++j) {
      const unsigned short* bp = Bt + (size_t)(n0 + 16 * j + low) * K + k0 + 8 * hi;
      Frag16 bfr;
      bfr.h[0] = *(const us8*)(bp);
      bfr.h[1] = *(const us8*)(bp + 16);
      acc[j] = __builtin_amdgcn_wmma_f32_16x16x32_bf16(
          false, af.v, false, bfr.v, (short)0, acc[j], false, false);
    }
  }

#pragma unroll
  for (int j = 0; j < 4; ++j) {
    int n = n0 + 16 * j + low;
#pragma unroll
    for (int r = 0; r < 8; ++r) {
      int m = m0 + r + 8 * hi;
      float v = acc[j][r];
      if (mode == 2) v += rowb[m] + colb[n] - NEGV * (1.0f - (float)maskp[n]);
      Co[(size_t)m * N + n] = v;
    }
  }
}

// ---------------------------------------------------------------------------
// LDS-tiled bf16 WMMA GEMM for the big projections (single batch):
//   C[m,n] = sum_k A[m,k]*Bt[n,k] (+bias[n])
// 256 threads = 8 waves. Block tile 128x128 (waves: 4 in M x 2 in N, each
// wave 32x64). K-slab 64, double-buffered LDS filled with async b128 copies.
// ---------------------------------------------------------------------------
#define KS   64
#define LSTR 72   // padded LDS row stride (elements)

__global__ void __launch_bounds__(256)
gemm_bf16_lds_kernel(const unsigned short* __restrict__ A,
                     const unsigned short* __restrict__ Bt,
                     float* __restrict__ Co,
                     int M, int N, int K,
                     const float* __restrict__ bias) {
  __shared__ unsigned short Asm[2][128 * LSTR];
  __shared__ unsigned short Bsm[2][128 * LSTR];

  int tid  = threadIdx.x;
  int lane = tid & 31;
  int wid  = tid >> 5;            // 0..7
  int wave_m = wid >> 1;          // 0..3  -> 32-row subtile
  int wave_n = wid & 1;           // 0..1  -> 64-col subtile
  int low = lane & 15, hi = lane >> 4;

  int n_blk = blockIdx.x * 128;
  int m_blk = blockIdx.y * 128;

  // cooperative copy mapping: 1024 chunks of 16B per (A|B) slab; 4 per thread
  int r0 = tid >> 3;              // 0..31
  int cc = (tid & 7) * 8;         // 0..56 (elements)

  v8f acc[2][4] = {};

  int nslabs = K >> 6;

#if defined(HAVE_ASYNC_LDS)
  auto issue_slab = [&](int k0, int buf) {
#pragma unroll
    for (int i = 0; i < 4; ++i) {
      int row = r0 + 32 * i;
      const unsigned short* ga = A  + (size_t)(m_blk + row) * K + k0 + cc;
      const unsigned short* gb = Bt + (size_t)(n_blk + row) * K + k0 + cc;
      __builtin_amdgcn_global_load_async_to_lds_b128(
          (gas_v4i*)(unsigned long long)(const void*)ga,
          (las_v4i*)(void*)&Asm[buf][row * LSTR + cc], 0, 0);
      __builtin_amdgcn_global_load_async_to_lds_b128(
          (gas_v4i*)(unsigned long long)(const void*)gb,
          (las_v4i*)(void*)&Bsm[buf][row * LSTR + cc], 0, 0);
    }
  };
#endif

  auto compute = [&](int buf) {
    const unsigned short* As = Asm[buf];
    const unsigned short* Bs = Bsm[buf];
#pragma unroll
    for (int kk = 0; kk < KS; kk += 32) {
      Frag16 af[2];
#pragma unroll
      for (int sub = 0; sub < 2; ++sub) {
        const unsigned short* ap =
            As + (size_t)(wave_m * 32 + sub * 16 + low) * LSTR + kk + 8 * hi;
        af[sub].h[0] = *(const us8*)(ap);
        af[sub].h[1] = *(const us8*)(ap + 16);
      }
#pragma unroll
      for (int j = 0; j < 4; ++j) {
        const unsigned short* bp =
            Bs + (size_t)(wave_n * 64 + j * 16 + low) * LSTR + kk + 8 * hi;
        Frag16 bfr;
        bfr.h[0] = *(const us8*)(bp);
        bfr.h[1] = *(const us8*)(bp + 16);
        acc[0][j] = __builtin_amdgcn_wmma_f32_16x16x32_bf16(
            false, af[0].v, false, bfr.v, (short)0, acc[0][j], false, false);
        acc[1][j] = __builtin_amdgcn_wmma_f32_16x16x32_bf16(
            false, af[1].v, false, bfr.v, (short)0, acc[1][j], false, false);
      }
    }
  };

#if defined(HAVE_ASYNC_LDS)
  issue_slab(0, 0);
  for (int s = 0; s < nslabs; ++s) {
    int cur = s & 1;
    if (s + 1 < nslabs) {
      issue_slab((s + 1) << 6, cur ^ 1);
      WAITASYNC(8);           // slab `cur` landed; 8 ops of next still in flight
    } else {
      WAITASYNC(0);
    }
    __syncthreads();          // all waves' copies for `cur` visible
    compute(cur);
    __syncthreads();          // all waves done reading `cur` before reuse
  }
#else
  for (int s = 0; s < nslabs; ++s) {
    int k0 = s << 6;
    us8 ra[4], rb[4];
#pragma unroll
    for (int i = 0; i < 4; ++i) {
      int row = r0 + 32 * i;
      ra[i] = *(const us8*)(A  + (size_t)(m_blk + row) * K + k0 + cc);
      rb[i] = *(const us8*)(Bt + (size_t)(n_blk + row) * K + k0 + cc);
    }
    __syncthreads();          // previous slab fully consumed
#pragma unroll
    for (int i = 0; i < 4; ++i) {
      int row = r0 + 32 * i;
      *(us8*)&Asm[0][row * LSTR + cc] = ra[i];
      *(us8*)&Bsm[0][row * LSTR + cc] = rb[i];
    }
    __syncthreads();
    compute(0);
  }
#endif

#pragma unroll
  for (int sub = 0; sub < 2; ++sub) {
#pragma unroll
    for (int j = 0; j < 4; ++j) {
      int n = n_blk + wave_n * 64 + j * 16 + low;
#pragma unroll
      for (int r = 0; r < 8; ++r) {
        int m = m_blk + wave_m * 32 + sub * 16 + r + 8 * hi;
        float v = acc[sub][j][r];
        if (bias) v += bias[n];
        Co[(size_t)m * N + n] = v;
      }
    }
  }
}

// ---------------------------------------------------------------------------
// Small helper kernels
// ---------------------------------------------------------------------------
__global__ void rowdot_kernel(const float* __restrict__ X, const float* __restrict__ w,
                              float* __restrict__ out, int rows, int D) {
  int wid  = (int)(((size_t)blockIdx.x * blockDim.x + threadIdx.x) >> 5);
  int lane = threadIdx.x & 31;
  if (wid >= rows) return;
  const float* xr = X + (size_t)wid * D;
  float s = 0.f;
  for (int d = lane; d < D; d += 32) s += xr[d] * w[d];
  for (int off = 16; off; off >>= 1) s += __shfl_down(s, off, 32);
  if (lane == 0) out[wid] = s;
}

__global__ void cast_scale_kernel(const float* __restrict__ X, const float* __restrict__ scale,
                                  unsigned short* __restrict__ Y, long long n, int dmask) {
  long long i = (long long)blockIdx.x * blockDim.x + threadIdx.x;
  if (i >= n) return;
  float v = X[i];
  if (scale) v *= scale[(int)(i & dmask)];
  Y[i] = f2bf(v);
}

__global__ void transpose_cast_kernel(const float* __restrict__ mem,
                                      unsigned short* __restrict__ memT, long long n) {
  long long i = (long long)blockIdx.x * blockDim.x + threadIdx.x;
  if (i >= n) return;
  int q = (int)(i % QQ);
  int d = (int)((i / QQ) % DD);
  int b = (int)(i / ((long long)QQ * DD));
  memT[i] = f2bf(mem[((size_t)b * QQ + q) * DD + d]);
}

__global__ void softmax_q_kernel(const float* __restrict__ S, unsigned short* __restrict__ Pbf,
                                 float* __restrict__ rowmax, int rows) {
  int wid  = (int)(((size_t)blockIdx.x * blockDim.x + threadIdx.x) >> 5);
  int lane = threadIdx.x & 31;
  if (wid >= rows) return;
  const float* sr = S + (size_t)wid * QQ;
  float a0 = sr[lane], a1 = sr[lane + 32];
  float m = fmaxf(a0, a1);
  for (int off = 16; off; off >>= 1) m = fmaxf(m, __shfl_xor(m, off, 32));
  float e0 = __expf(a0 - m), e1 = __expf(a1 - m);
  float s = e0 + e1;
  for (int off = 16; off; off >>= 1) s += __shfl_xor(s, off, 32);
  float inv = 1.0f / s;
  Pbf[(size_t)wid * QQ + lane]      = f2bf(e0 * inv);
  Pbf[(size_t)wid * QQ + lane + 32] = f2bf(e1 * inv);
  if (lane == 0) rowmax[wid] = m;
}

__global__ void softmax_c_kernel(const float* __restrict__ rowmax, float* __restrict__ wtwo) {
  __shared__ float red[256];
  int b = blockIdx.x, tid = threadIdx.x;
  const float* r = rowmax + (size_t)b * CC;
  float m = -3.4e38f;
  for (int c = tid; c < CC; c += 256) m = fmaxf(m, r[c]);
  red[tid] = m; __syncthreads();
  for (int o = 128; o; o >>= 1) { if (tid < o) red[tid] = fmaxf(red[tid], red[tid + o]); __syncthreads(); }
  float mm = red[0]; __syncthreads();
  float s = 0.f;
  for (int c = tid; c < CC; c += 256) s += __expf(r[c] - mm);
  red[tid] = s; __syncthreads();
  for (int o = 128; o; o >>= 1) { if (tid < o) red[tid] += red[tid + o]; __syncthreads(); }
  float inv = 1.0f / red[0];
  for (int c = tid; c < CC; c += 256) wtwo[(size_t)b * CC + c] = __expf(r[c] - mm) * inv;
}

__global__ void outtwo_kernel(const float* __restrict__ wtwo, const float* __restrict__ X,
                              float* __restrict__ otwo) {
  int d = blockIdx.x * blockDim.x + threadIdx.x;
  int b = blockIdx.y;
  if (d >= DD) return;
  const float* xb = X + (size_t)b * CC * DD;
  const float* wb = wtwo + (size_t)b * CC;
  float s = 0.f;
  for (int c = 0; c < CC; ++c) s += wb[c] * xb[(size_t)c * DD + d];
  otwo[(size_t)b * DD + d] = s;
}

__global__ void concat_cast_kernel(const float* __restrict__ X, const float* __restrict__ O1,
                                   const float* __restrict__ otwo, unsigned short* __restrict__ Abf,
                                   long long n) {
  long long i = (long long)blockIdx.x * blockDim.x + threadIdx.x;
  if (i >= n) return;
  int d = (int)(i & (DD - 1));
  long long row = i >> 10;
  int b = (int)(row / CC);
  float x = X[i], o = O1[i], t = otwo[(size_t)b * DD + d];
  size_t base = (size_t)row * (4 * DD) + d;
  Abf[base]          = f2bf(x);
  Abf[base + DD]     = f2bf(o);
  Abf[base + 2 * DD] = f2bf(x * o);
  Abf[base + 3 * DD] = f2bf(t * o);
}

// ---------------------------------------------------------------------------
// Host orchestration
// ---------------------------------------------------------------------------
static inline size_t al256(size_t x) { return (x + 255) & ~(size_t)255; }

extern "C" void kernel_launch(void* const* d_in, const int* in_sizes, int n_in,
                              void* d_out, int out_size, void* d_ws, size_t ws_size,
                              hipStream_t stream) {
  (void)in_sizes; (void)n_in; (void)out_size; (void)ws_size;
  const float* ctx   = (const float*)d_in[0];
  const float* mem1  = (const float*)d_in[1];
  const float* mem2  = (const float*)d_in[2];
  const int*   mask1 = (const int*)d_in[3];
  const int*   mask2 = (const int*)d_in[4];
  const float* w_in1 = (const float*)d_in[5];
  const float* w_mem1= (const float*)d_in[6];
  const float* scale1= (const float*)d_in[7];
  const float* w_out1= (const float*)d_in[8];
  const float* w_in2 = (const float*)d_in[9];
  const float* w_mem2= (const float*)d_in[10];
  const float* scale2= (const float*)d_in[11];
  const float* w_out2= (const float*)d_in[12];
  const float* w_map = (const float*)d_in[13];
  const float* b_map = (const float*)d_in[14];
  float* out = (float*)d_out;

  char* p = (char*)d_ws; size_t off = 0;
  auto take = [&](size_t bytes) { void* r = p + off; off = al256(off + bytes); return r; };
  float*          xbuf   = (float*)take((size_t)BB * CC * DD * 4);
  float*          o1buf  = (float*)take((size_t)BB * CC * DD * 4);
  float*          Sbuf   = (float*)take((size_t)BB * CC * QQ * 4);
  float*          rdC    = (float*)take((size_t)BB * CC * 4);
  float*          rdQ    = (float*)take((size_t)BB * QQ * 4);
  float*          rmax   = (float*)take((size_t)BB * CC * 4);
  float*          wtwo   = (float*)take((size_t)BB * CC * 4);
  float*          otwo   = (float*)take((size_t)BB * DD * 4);
  unsigned short* Pbf    = (unsigned short*)take((size_t)BB * CC * QQ * 2);
  unsigned short* xsbf   = (unsigned short*)take((size_t)BB * CC * DD * 2);  // reused as X2bf
  unsigned short* membf  = (unsigned short*)take((size_t)BB * QQ * DD * 2);
  unsigned short* memTbf = (unsigned short*)take((size_t)BB * DD * QQ * 2);
  unsigned short* Abf    = (unsigned short*)take((size_t)BB * CC * 4 * DD * 2);
  unsigned short* Wbf    = (unsigned short*)take((size_t)4 * DD * DD * 2);

  const long long nX   = (long long)BB * CC * DD;   // 8388608
  const long long nMem = (long long)BB * QQ * DD;   // 262144

  auto run_pass = [&](const float* x, const float* mem, const int* mask,
                      const float* w_in, const float* w_mem, const float* scale,
                      const float* w_out, float* xout) {
    rowdot_kernel<<<(BB * CC * 32) / 256, 256, 0, stream>>>(x, w_in, rdC, BB * CC, DD);
    rowdot_kernel<<<(BB * QQ * 32) / 256, 256, 0, stream>>>(mem, w_mem, rdQ, BB * QQ, DD);
    cast_scale_kernel<<<(int)(nX / 256), 256, 0, stream>>>(x, scale, xsbf, nX, DD - 1);
    cast_scale_kernel<<<(int)(nMem / 256), 256, 0, stream>>>(mem, nullptr, membf, nMem, 0);
    transpose_cast_kernel<<<(int)(nMem / 256), 256, 0, stream>>>(mem, memTbf, nMem);
    cast_scale_kernel<<<(int)((long long)4 * DD * DD / 256), 256, 0, stream>>>(
        w_out, nullptr, Wbf, (long long)4 * DD * DD, 0);
    // S = (x*scale) @ mem^T + rowdot + memdot - NEG*(1-mask)
    {
      int waves = BB * (CC / 16) * (QQ / 64);
      gemm_bf16_kernel<<<(waves * 32 + 255) / 256, 256, 0, stream>>>(
          xsbf, membf, Sbuf, CC, QQ, DD,
          (long long)CC * DD, (long long)QQ * DD, (long long)CC * QQ,
          BB, 2, rdC, rdQ, mask);
    }
    softmax_q_kernel<<<(BB * CC * 32) / 256, 256, 0, stream>>>(Sbuf, Pbf, rmax, BB * CC);
    softmax_c_kernel<<<BB, 256, 0, stream>>>(rmax, wtwo);
    outtwo_kernel<<<dim3(DD / 256, BB), 256, 0, stream>>>(wtwo, x, otwo);
    // output_one = P @ mem
    {
      int waves = BB * (CC / 16) * (DD / 64);
      gemm_bf16_kernel<<<(waves * 32 + 255) / 256, 256, 0, stream>>>(
          Pbf, memTbf, o1buf, CC, DD, QQ,
          (long long)CC * QQ, (long long)DD * QQ, (long long)CC * DD,
          BB, 0, nullptr, nullptr, nullptr);
    }
    concat_cast_kernel<<<(int)(nX / 256), 256, 0, stream>>>(x, o1buf, otwo, Abf, nX);
    // dominant GEMM: xout = Abf @ w_out^T   (LDS-tiled, async copies)
    gemm_bf16_lds_kernel<<<dim3(DD / 128, (BB * CC) / 128), 256, 0, stream>>>(
        Abf, Wbf, xout, BB * CC, DD, 4 * DD, nullptr);
  };

  run_pass(ctx,  mem1, mask1, w_in1, w_mem1, scale1, w_out1, xbuf);
  run_pass(xbuf, mem2, mask2, w_in2, w_mem2, scale2, w_out2, xbuf);

  // final: out = x2 @ w_map^T + b_map
  cast_scale_kernel<<<(int)(nX / 256), 256, 0, stream>>>(xbuf, nullptr, xsbf, nX, 0);
  cast_scale_kernel<<<(int)((long long)2 * DD * DD / 256), 256, 0, stream>>>(
      w_map, nullptr, Wbf, (long long)2 * DD * DD, 0);
  gemm_bf16_lds_kernel<<<dim3((2 * DD) / 128, (BB * CC) / 128), 256, 0, stream>>>(
      xsbf, Wbf, out, BB * CC, 2 * DD, DD, b_map);
}
